// BatchGRU_4844723110205
// MI455X (gfx1250) — matrix operand order repaired and checked by reading, not compiled, not run on hardware
//
#include <hip/hip_runtime.h>
#include <hip/hip_bf16.h>
#include <math.h>

// ---------------------------------------------------------------------------
// BatchGRU on MI455X (gfx1250, wave32, WMMA bf16 16x16x32, f32 accumulate)
//
// Shapes: H=300, B=2048, SEG=48, N=98304.  All matmuls via
// v_wmma_f32_16x16x32_bf16.  Hidden padded 300->320 (K), 900->912 (N).
// Each wave processes THREE 16-col tiles at once (57 = 19*3).  The k-loop is
// only partially unrolled (x2) so the per-k A-fragment LDS loads depend on a
// runtime induction variable: LICM cannot hoist them out of the tile loop and
// re-create the 80-VGPR A array that previously spilled to scratch.
// ---------------------------------------------------------------------------

#define Hh    300
#define HP    304          // padded f32 hidden stride (LDS)
#define KP    320          // padded bf16 K stride (10 k-tiles of 32)
#define G3    900          // 3*H
#define G3P   912          // padded, 57 n-tiles of 16
#define BATCH 2048
#define SEG   48
#define NTOT  (BATCH * SEG)
#define NT16  (G3P / 16)   // 57
#define NTRI  (NT16 / 3)   // 19 tile-triples
#define KT32  (KP / 32)    // 10

typedef __attribute__((ext_vector_type(16))) __bf16 v16bf;
typedef __attribute__((ext_vector_type(8)))  float  v8f;

struct Frag32 { float4 lo, hi; };   // 32 bytes == one v16bf fragment

static __device__ __forceinline__ unsigned short f2bf(float f) {
    // round-to-nearest-even f32 -> bf16
    unsigned u = __float_as_uint(f);
    unsigned r = u + 0x7FFFu + ((u >> 16) & 1u);
    return (unsigned short)(r >> 16);
}
static __device__ __forceinline__ float bf2f(unsigned short h) {
    return __uint_as_float(((unsigned)h) << 16);
}

// A-matrix 16x32 bf16 fragment: lanes 0-15 hold K {k0..k0+15},
// lanes 16-31 hold K {k0+16..k0+31}, row m = lane&15.
static __device__ __forceinline__ v16bf load_a_frag(const unsigned short* rowBase,
                                                    int k0, int hi8) {
    Frag32 f;
    f.lo = *reinterpret_cast<const float4*>(rowBase + k0 + hi8 * 16);
    f.hi = *reinterpret_cast<const float4*>(rowBase + k0 + hi8 * 16 + 8);
    return __builtin_bit_cast(v16bf, f);
}

// B-matrix 32x16 bf16 fragment: lane holds col n=lane&15, K run k0+(lane>>4)*16 .. +15,
// which for C = A * W^T is 16 *contiguous* bf16 from row n of W.
static __device__ __forceinline__ v16bf load_b_frag(const unsigned short* p) {
    Frag32 f;
    f.lo = *reinterpret_cast<const float4*>(p);
    f.hi = *reinterpret_cast<const float4*>(p + 8);
    return __builtin_bit_cast(v16bf, f);
}

static __device__ __forceinline__ v8f wmma_bf16(v16bf a, v16bf b, v8f c) {
    return __builtin_amdgcn_wmma_f32_16x16x32_bf16(false, a, false, b,
                                                   (short)0, c, false, false);
}

// ---------------------------------------------------------------------------
// K0: convert the 4 weight matrices [900,300] f32 -> padded bf16 [912,320]
//     order in wB: whh_f, whh_b, wih_f, wih_b
// ---------------------------------------------------------------------------
__global__ void k_wconv(const float* __restrict__ whhf, const float* __restrict__ whhb,
                        const float* __restrict__ wihf, const float* __restrict__ wihb,
                        unsigned short* __restrict__ wB) {
    const size_t per = (size_t)G3P * KP;
    size_t idx = (size_t)blockIdx.x * blockDim.x + threadIdx.x;
    if (idx >= 4 * per) return;
    int mat = (int)(idx / per);
    size_t rem = idx - (size_t)mat * per;
    int r = (int)(rem / KP);
    int c = (int)(rem - (size_t)r * KP);
    float v = 0.f;
    if (r < G3 && c < Hh) {
        const float* src = (mat == 0) ? whhf : (mat == 1) ? whhb
                         : (mat == 2) ? wihf : wihb;
        v = src[(size_t)r * Hh + c];
    }
    wB[idx] = f2bf(v);
}

// ---------------------------------------------------------------------------
// K1: message = relu(x + bias) -> bf16 padded [N,320];  h0 = segment max(x) [B,304] f32
//     one block per molecule (48 rows), single pass over x.
// ---------------------------------------------------------------------------
__global__ void k_prep(const float* __restrict__ x, const float* __restrict__ bias,
                       unsigned short* __restrict__ msgB, float* __restrict__ h0F) {
    const int b = blockIdx.x;
    for (int j = threadIdx.x; j < KP; j += blockDim.x) {
        if (j < Hh) {
            float bj = bias[j];
            float mx = -3.4e38f;
            for (int t = 0; t < SEG; ++t) {
                float v = x[((size_t)b * SEG + t) * Hh + j];
                mx = fmaxf(mx, v);
                msgB[((size_t)b * SEG + t) * KP + j] = f2bf(fmaxf(v + bj, 0.f));
            }
            h0F[(size_t)b * HP + j] = mx;
        } else {
            for (int t = 0; t < SEG; ++t)
                msgB[((size_t)b * SEG + t) * KP + j] = 0;   // K padding
            if (j < HP) h0F[(size_t)b * HP + j] = 0.f;
        }
    }
}

// ---------------------------------------------------------------------------
// K2: input projection xp = msg @ w_ih^T + b_ih  -> bf16 [N,912] per direction
//     grid (N/16, 2).  Block stages its 16x320 msg tile in LDS; 8 waves split
//     the 19 tile-triples; per k-step: 2 LDS + 6 global b128 loads, 3 WMMAs.
// ---------------------------------------------------------------------------
__global__ void k_xproj(const unsigned short* __restrict__ msgB,
                        const unsigned short* __restrict__ wB,
                        const float* __restrict__ bih_f, const float* __restrict__ bih_b,
                        unsigned short* __restrict__ xpB) {
    extern __shared__ unsigned short sA[];                 // 16*KP bf16 = 10240 B
    const int dir = blockIdx.y;
    const unsigned short* wih = wB + (size_t)(2 + dir) * (G3P * KP);
    const float* bih = dir ? bih_b : bih_f;
    unsigned short* xp = xpB + (size_t)dir * NTOT * G3P;
    const int mtile = blockIdx.x;

    // stage contiguous 16-row msg tile into LDS (640 x 16B)
    const uint4* src = reinterpret_cast<const uint4*>(msgB + (size_t)mtile * 16 * KP);
    uint4* dst = reinterpret_cast<uint4*>(sA);
    for (int i = threadIdx.x; i < (16 * KP * 2) / 16; i += blockDim.x) dst[i] = src[i];
    __syncthreads();

    const int lane = threadIdx.x & 31, wave = threadIdx.x >> 5;
    const int n16 = lane & 15, hi8 = lane >> 4;
    const unsigned short* aRow = sA + n16 * KP;

    for (int p = wave; p < NTRI; p += 8) {
        const int nt0 = p * 3;
        const unsigned short* w0 = wih + (size_t)(nt0 * 16 + n16) * KP + hi8 * 16;
        v8f acc0 = {}, acc1 = {}, acc2 = {};
#pragma unroll 2
        for (int kt = 0; kt < KT32; ++kt) {      // partial unroll: A-load stays in-loop
            v16bf a  = load_a_frag(aRow, kt * 32, hi8);          // LDS, low latency
            v16bf b0 = load_b_frag(w0 + kt * 32);                // row nt0*16+n16
            v16bf b1 = load_b_frag(w0 + 16 * KP + kt * 32);      // +16 rows
            v16bf b2 = load_b_frag(w0 + 32 * KP + kt * 32);      // +32 rows
            acc0 = wmma_bf16(a, b0, acc0);
            acc1 = wmma_bf16(a, b1, acc1);
            acc2 = wmma_bf16(a, b2, acc2);
        }
        const size_t rowTop = (size_t)mtile * 16 + hi8 * 8;
#pragma unroll
        for (int q = 0; q < 3; ++q) {
            const int col = (nt0 + q) * 16 + n16;
            const float bb = (col < G3) ? bih[col] : 0.f;
            const v8f acc = (q == 0) ? acc0 : (q == 1) ? acc1 : acc2;
            size_t rbase = rowTop * G3P + col;
#pragma unroll
            for (int r = 0; r < 8; ++r)
                xp[rbase + (size_t)r * G3P] = f2bf(acc[r] + bb);
        }
    }
}

// ---------------------------------------------------------------------------
// K3: persistent bidirectional GRU.  grid (B/16, 2 dirs), 256 threads.
//     Block owns 16 molecules; h lives in LDS across all 48 steps (the
//     recurrence never mixes batch rows, so no inter-block sync is needed).
//     Per step: hp = h_bf16 @ w_hh^T (WMMA, hp -> LDS), then fused gates.
// ---------------------------------------------------------------------------
__global__ void k_gru(const unsigned short* __restrict__ wB,
                      const float* __restrict__ h0F,
                      const unsigned short* __restrict__ xpB,
                      const float* __restrict__ bhh_f, const float* __restrict__ bhh_b,
                      float* __restrict__ out) {
    extern __shared__ char smem[];
    float* hS  = reinterpret_cast<float*>(smem);                    // 16*HP f32
    float* hpS = hS + 16 * HP;                                      // 16*G3P f32
    unsigned short* hB = reinterpret_cast<unsigned short*>(hpS + 16 * G3P); // 16*KP bf16

    const int dir = blockIdx.y;
    const unsigned short* whh = wB + (size_t)dir * (G3P * KP);
    const unsigned short* xp  = xpB + (size_t)dir * NTOT * G3P;
    const float* bhh = dir ? bhh_b : bhh_f;
    const int bBase = blockIdx.x * 16;
    const int tid = threadIdx.x;

    // h := h0 (f32 + bf16 copies, K padding zeroed once)
    for (int idx = tid; idx < 16 * KP; idx += 256) {
        int m = idx / KP, j = idx - m * KP;
        float v = (j < Hh) ? h0F[(size_t)(bBase + m) * HP + j] : 0.f;
        hB[m * KP + j] = f2bf(v);
        if (j < HP) hS[m * HP + j] = v;
    }
    __syncthreads();

    const int lane = tid & 31, wave = tid >> 5;
    const int n16 = lane & 15, hi8 = lane >> 4;
    const unsigned short* aRow = hB + n16 * KP;

    for (int s = 0; s < SEG; ++s) {
        const int t = dir ? (SEG - 1 - s) : s;

        // ---- hp = h @ w_hh^T  (M=16, N=912, K=320), 3 tiles per wave-iter ----
        for (int p = wave; p < NTRI; p += 8) {
            const int nt0 = p * 3;
            const unsigned short* w0 = whh + (size_t)(nt0 * 16 + n16) * KP + hi8 * 16;
            v8f acc0 = {}, acc1 = {}, acc2 = {};
#pragma unroll 2
            for (int kt = 0; kt < KT32; ++kt) {  // partial unroll: A-load stays in-loop
                v16bf a  = load_a_frag(aRow, kt * 32, hi8);      // LDS h state
                v16bf b0 = load_b_frag(w0 + kt * 32);
                v16bf b1 = load_b_frag(w0 + 16 * KP + kt * 32);
                v16bf b2 = load_b_frag(w0 + 32 * KP + kt * 32);
                acc0 = wmma_bf16(a, b0, acc0);
                acc1 = wmma_bf16(a, b1, acc1);
                acc2 = wmma_bf16(a, b2, acc2);
            }
#pragma unroll
            for (int q = 0; q < 3; ++q) {
                const int col = (nt0 + q) * 16 + n16;
                const v8f acc = (q == 0) ? acc0 : (q == 1) ? acc1 : acc2;
                if (col < G3) {
#pragma unroll
                    for (int r = 0; r < 8; ++r)
                        hpS[(hi8 * 8 + r) * G3P + col] = acc[r];
                }
            }
        }
        __syncthreads();

        // ---- fused gates: r,z,n + state update + output store ----
        for (int idx = tid; idx < 16 * Hh; idx += 256) {
            int m = idx / Hh, j = idx - m * Hh;
            size_t g = (size_t)(bBase + m) * SEG + t;
            const unsigned short* xprow = xp + g * G3P;
            float xr = bf2f(xprow[j]);
            float xz = bf2f(xprow[Hh + j]);
            float xn = bf2f(xprow[2 * Hh + j]);
            float hr = hpS[m * G3P + j]          + bhh[j];
            float hz = hpS[m * G3P + Hh + j]     + bhh[Hh + j];
            float hn = hpS[m * G3P + 2 * Hh + j] + bhh[2 * Hh + j];
            float r = 1.f / (1.f + __expf(-(xr + hr)));
            float z = 1.f / (1.f + __expf(-(xz + hz)));
            float n = tanhf(xn + r * hn);
            float hOld = hS[m * HP + j];
            float hNew = (1.f - z) * n + z * hOld;
            hS[m * HP + j] = hNew;
            hB[m * KP + j] = f2bf(hNew);
            out[g * (2 * Hh) + dir * Hh + j] = hNew;
        }
        __syncthreads();
    }
}

// ---------------------------------------------------------------------------
extern "C" void kernel_launch(void* const* d_in, const int* in_sizes, int n_in,
                              void* d_out, int out_size, void* d_ws, size_t ws_size,
                              hipStream_t stream) {
    (void)in_sizes; (void)n_in; (void)out_size; (void)ws_size;
    const float* x     = (const float*)d_in[0];
    // d_in[1]=batch, d_in[2]=num_moles, d_in[3]=max_len: implied by fixed shapes
    const float* bias  = (const float*)d_in[4];
    const float* wih_f = (const float*)d_in[5];
    const float* whh_f = (const float*)d_in[6];
    const float* bih_f = (const float*)d_in[7];
    const float* bhh_f = (const float*)d_in[8];
    const float* wih_b = (const float*)d_in[9];
    const float* whh_b = (const float*)d_in[10];
    const float* bih_b = (const float*)d_in[11];
    const float* bhh_b = (const float*)d_in[12];
    float* out = (float*)d_out;

    // workspace carve-up (16B aligned)
    char* ws = (char*)d_ws;
    unsigned short* msgB = (unsigned short*)ws;                    // N*KP bf16
    size_t off = (size_t)NTOT * KP * 2;
    float* h0F = (float*)(ws + off);                               // B*HP f32
    off += (size_t)BATCH * HP * 4;
    unsigned short* wB = (unsigned short*)(ws + off);              // 4 * G3P*KP bf16
    off += (size_t)4 * G3P * KP * 2;
    unsigned short* xpB = (unsigned short*)(ws + off);             // 2 * N*G3P bf16

    // K0: weight convert/pad
    {
        size_t total = (size_t)4 * G3P * KP;
        int blocks = (int)((total + 255) / 256);
        k_wconv<<<blocks, 256, 0, stream>>>(whh_f, whh_b, wih_f, wih_b, wB);
    }
    // K1: relu(x+bias) -> bf16 msg, segment max -> h0
    k_prep<<<BATCH, 256, 0, stream>>>(x, bias, msgB, h0F);
    // K2: input projections for both directions
    {
        dim3 grid(NTOT / 16, 2);
        k_xproj<<<grid, 256, 16 * KP * 2, stream>>>(msgB, wB, bih_f, bih_b, xpB);
    }
    // K3: persistent recurrent GRU, both directions
    {
        dim3 grid(BATCH / 16, 2);
        size_t shmem = 16 * HP * 4 + 16 * G3P * 4 + 16 * KP * 2;   // 88,064 B
        k_gru<<<grid, 256, shmem, stream>>>(wB, h0F, xpB, bhh_f, bhh_b, out);
    }
}
